// StateMixer_7791070675547
// MI455X (gfx1250) — compile-verified, matrix operand order checked by reading, not run on hardware
//
#include <hip/hip_runtime.h>
#include <hip/hip_bf16.h>

typedef __bf16 bf16;
typedef __attribute__((ext_vector_type(8)))  __bf16 v8bf;
typedef __attribute__((ext_vector_type(16))) __bf16 v16bf;
typedef __attribute__((ext_vector_type(8)))  float  v8f;

#define CDIM  128
#define GGDIM 256
#define FS    136   // f32 LDS stride for async x tiles (544B rows: 16B aligned, even bank spread)
#define XSS   130   // f32 LDS stride for xs tile

// lane <- lane ^ m exchange within wave32 (group-of-32 ds_swizzle: and=0x1f, or=0, xor=m)
#define SWZ_ADD(v, imm)                                                            \
  (v) += __int_as_float(__builtin_amdgcn_ds_swizzle(__float_as_int(v), (imm)))

static __device__ __forceinline__ v16bf cat8(v8bf lo, v8bf hi) {
  return __builtin_shufflevector(lo, hi, 0,1,2,3,4,5,6,7,8,9,10,11,12,13,14,15);
}

static __device__ __forceinline__ v16bf cvt16(const float* p) {
  // 8 consecutive f32 at p, 8 more at p+16 (WMMA bf16 A/B half-lane K layout)
  const float4 a0 = *(const float4*)(p);
  const float4 a1 = *(const float4*)(p + 4);
  const float4 a2 = *(const float4*)(p + 16);
  const float4 a3 = *(const float4*)(p + 20);
  v16bf v;
  v[0]  = (bf16)a0.x; v[1]  = (bf16)a0.y; v[2]  = (bf16)a0.z; v[3]  = (bf16)a0.w;
  v[4]  = (bf16)a1.x; v[5]  = (bf16)a1.y; v[6]  = (bf16)a1.z; v[7]  = (bf16)a1.w;
  v[8]  = (bf16)a2.x; v[9]  = (bf16)a2.y; v[10] = (bf16)a2.z; v[11] = (bf16)a2.w;
  v[12] = (bf16)a3.x; v[13] = (bf16)a3.y; v[14] = (bf16)a3.z; v[15] = (bf16)a3.w;
  return v;
}

// ---------------------------------------------------------------------------
// xd = tok @ Wr + br   (per type, [C] vector) — tiny, exact fp32
// ---------------------------------------------------------------------------
__global__ void xd_kernel(const float* __restrict__ tok, const float* __restrict__ Wr,
                          const float* __restrict__ br, float* __restrict__ xd) {
  const int j = threadIdx.x;
  float s = br[j];
  for (int k = 0; k < CDIM; ++k) s = fmaf(tok[k], Wr[k * CDIM + j], s);
  xd[j] = s;
}

// ---------------------------------------------------------------------------
// Fused GATv2-to-global for one node type. One block (8 waves) per graph.
// x tiles stream HBM->LDS with global_load_async_to_lds_b128 (triple buffered,
// ASYNCcnt-paced); xs = x@Wl+bl via v_wmma_f32_16x16x32_bf16; attention logits
// + online softmax keep xs in LDS so it never touches HBM.
// ---------------------------------------------------------------------------
__global__ __launch_bounds__(256) void gat_kernel(
    const float* __restrict__ x, const float* __restrict__ Wl,
    const float* __restrict__ bl, const float* __restrict__ att,
    const float* __restrict__ xd, const float* __restrict__ bo,
    float* __restrict__ out, int N, int G)
{
  __shared__ float xbuf[3 * 16 * FS];   // triple-buffered f32 x tiles
  __shared__ float xs_tile[16 * XSS];
  __shared__ float e_tile[16];
  __shared__ float comb[CDIM];
  __shared__ float combd[CDIM];

  const int tid  = threadIdx.x;
  const int lane = tid & 31;
  const int wave = tid >> 5;      // 0..7 : output column tile
  const int half = lane >> 4;     // 0/1  : WMMA half-wave
  const int l16  = lane & 15;

  const int g     = blockIdx.x;
  const int seg_s = (int)(((long long)g       * N + G - 1) / G);
  const int seg_e = (int)(((long long)(g + 1) * N + G - 1) / G);

  const int   bcol  = wave * 16 + l16;        // this lane's output column
  const float bl_b  = bl[bcol];
  const float xd_b  = xd[bcol];
  const float att_b = att[bcol];

  // Preload this wave's B slice (Wl columns [16w,16w+15], all K=128): 32 VGPRs.
  v16bf breg[4];
  #pragma unroll
  for (int kk = 0; kk < 4; ++kk) {
    #pragma unroll
    for (int j = 0; j < 8; ++j) {
      breg[kk][j]     = (bf16)Wl[(kk * 32 +      8 * half + j) * CDIM + bcol];
      breg[kk][j + 8] = (bf16)Wl[(kk * 32 + 16 + 8 * half + j) * CDIM + bcol];
    }
  }

  const int n      = seg_e - seg_s;
  const int ntiles = (n + 15) >> 4;

  // Issue one 16x128 f32 tile as 512 async b128 transfers (2 per thread).
  // Rows past the segment end are clamped to the last valid node: data stays
  // finite (exact duplicate of a valid row) and gets zero softmax weight.
  auto issue_tile = [&](int t, int slot) {
    const int row0 = seg_s + t * 16;
    const int bsel = slot * 16 * FS;
    #pragma unroll
    for (int u = 0; u < 2; ++u) {
      const int idx = tid + u * 256;
      const int r   = idx >> 5;          // row 0..15
      const int c4  = (idx & 31) * 4;    // float offset 0..124
      int node = row0 + r;
      if (node >= seg_e) node = seg_e - 1;
      const float*   gp  = x + (size_t)node * CDIM + c4;
      const unsigned lds = (unsigned)(uintptr_t)&xbuf[bsel + r * FS + c4];
      asm volatile("global_load_async_to_lds_b128 %0, %1, off"
                   :: "v"(lds), "v"(gp) : "memory");
    }
  };

  // Online-softmax state: each thread owns (col = tid&127, rows myrow0..+7).
  // den is tracked per row-half (same rescale chain -> halves add linearly).
  const int mycol  = tid & 127;
  const int myrow0 = (tid >> 7) * 8;
  float m_run   = -__builtin_inff();
  float den     = 0.f;
  float acc_out = 0.f;

  issue_tile(0, 0);
  if (ntiles > 1) issue_tile(1, 1);

  for (int t = 0; t < ntiles; ++t) {
    const int row0 = seg_s + t * 16;

    // Keep 2 tile batches in flight; each batch is exactly 2 async ops per
    // wave, so tile t has landed once ASYNCcnt <= 2 * (#newer batches).
    if (t + 2 < ntiles) issue_tile(t + 2, (t + 2) % 3);
    const int newer = ntiles - 1 - t;
    if (newer >= 2)      asm volatile("s_wait_asynccnt 4" ::: "memory");
    else if (newer == 1) asm volatile("s_wait_asynccnt 2" ::: "memory");
    else                 asm volatile("s_wait_asynccnt 0" ::: "memory");

    if (tid < 16) e_tile[tid] = 0.f;
    __syncthreads();

    // Load all four A fragments (f32 LDS -> bf16 regs), then 4 chained WMMAs.
    const float* cur  = xbuf + (t % 3) * 16 * FS;
    const float* arow = cur + l16 * FS + 8 * half;
    v16bf av0 = cvt16(arow);
    v16bf av1 = cvt16(arow + 32);
    v16bf av2 = cvt16(arow + 64);
    v16bf av3 = cvt16(arow + 96);
    v8f accv = {0.f, 0.f, 0.f, 0.f, 0.f, 0.f, 0.f, 0.f};
    accv = __builtin_amdgcn_wmma_f32_16x16x32_bf16(false, av0, false, breg[0], (short)0, accv, false, false);
    accv = __builtin_amdgcn_wmma_f32_16x16x32_bf16(false, av1, false, breg[1], (short)0, accv, false, false);
    accv = __builtin_amdgcn_wmma_f32_16x16x32_bf16(false, av2, false, breg[2], (short)0, accv, false, false);
    accv = __builtin_amdgcn_wmma_f32_16x16x32_bf16(false, av3, false, breg[3], (short)0, accv, false, false);

    // Write xs to LDS; per-column contribution to attention logit.
    float econ[8];
    #pragma unroll
    for (int r = 0; r < 8; ++r) {
      const int   row = half * 8 + r;
      const float xsv = accv[r] + bl_b;
      xs_tile[row * XSS + bcol] = xsv;
      const float h = xsv + xd_b;
      econ[r] = att_b * (h > 0.f ? h : 0.2f * h);
    }
    // XOR-exchange reduce over the 16 lanes of each half, one ds_add per row.
    #pragma unroll
    for (int r = 0; r < 8; ++r) {
      float v = econ[r];
      SWZ_ADD(v, 0x041f);   // xor 1
      SWZ_ADD(v, 0x081f);   // xor 2
      SWZ_ADD(v, 0x101f);   // xor 4
      SWZ_ADD(v, 0x201f);   // xor 8
      if (l16 == 0) atomicAdd(&e_tile[half * 8 + r], v);
    }
    __syncthreads();

    // Online softmax update. Padding rows duplicate a valid row, so the tile
    // max needs no masking; only this thread's 8 ex values are guarded.
    float tmax = e_tile[0];
    #pragma unroll
    for (int r = 1; r < 16; ++r) tmax = fmaxf(tmax, e_tile[r]);
    const float nm = fmaxf(m_run, tmax);
    const float sc = __expf(m_run - nm);   // 0 on first tile (m_run = -inf)
    m_run = nm;
    float dsum = 0.f;
    float a = acc_out * sc;
    #pragma unroll
    for (int r = 0; r < 8; ++r) {
      const int   row = myrow0 + r;
      const float er  = (row0 + row < seg_e) ? e_tile[row] : -__builtin_inff();
      const float ex  = __expf(er - nm);
      dsum += ex;
      a = fmaf(ex, xs_tile[row * XSS + mycol], a);
    }
    acc_out = a;
    den = den * sc + dsum;
    __syncthreads();   // xs/e consumed; next iter may overwrite + async-write bufs
  }

  // Combine the two row-half accumulators and denominators; normalize; + bo.
  if (tid >= 128) { comb[mycol] = acc_out; combd[mycol] = den; }
  __syncthreads();
  if (tid < 128) {
    const float total = acc_out + comb[tid];
    const float dtot  = den + combd[tid];
    out[(size_t)g * CDIM + tid] = total / dtot + bo[tid];
  }
}

// ---------------------------------------------------------------------------
// Mix MLP: h=[gop|gma|gagv] (G x 384) -> lrelu(h@W1+b1,.01) @ W2 + b2 (G x 256)
// One block per 16 graphs; both GEMMs via bf16 WMMA.
// ---------------------------------------------------------------------------
__global__ __launch_bounds__(256) void mix_kernel(
    const float* __restrict__ gop, const float* __restrict__ gma, const float* __restrict__ gagv,
    const float* __restrict__ W1, const float* __restrict__ b1,
    const float* __restrict__ W2, const float* __restrict__ b2,
    float* __restrict__ gfeat, int G)
{
  const int HS = 392, H1S = 264;      // bf16 strides: 16B-aligned rows, conflict-free
  __shared__ bf16 h_tile[16 * 392];
  __shared__ bf16 h1_tile[16 * 264];

  const int tid = threadIdx.x, lane = tid & 31, wave = tid >> 5;
  const int half = lane >> 4, l16 = lane & 15;
  const int g0 = blockIdx.x * 16;

  // Stage h tile (concat of the three 128-wide blocks), f32 -> bf16.
  for (int i = tid; i < 16 * 192; i += 256) {
    const int r  = i / 192;
    const int kp = (i % 192) * 2;
    const int g  = g0 + r;
    float v0 = 0.f, v1 = 0.f;
    if (g < G) {
      const float* src;
      if (kp < 128)      src = gop  + (size_t)g * 128 + kp;
      else if (kp < 256) src = gma  + (size_t)g * 128 + (kp - 128);
      else               src = gagv + (size_t)g * 128 + (kp - 256);
      v0 = src[0]; v1 = src[1];
    }
    h_tile[r * HS + kp]     = (bf16)v0;
    h_tile[r * HS + kp + 1] = (bf16)v1;
  }
  __syncthreads();

  // GEMM1: 16x384 @ 384x256, each wave does two 16-col tiles.
  #pragma unroll
  for (int c = 0; c < 2; ++c) {
    const int col = (wave + c * 8) * 16 + l16;
    v8f acc = {0.f, 0.f, 0.f, 0.f, 0.f, 0.f, 0.f, 0.f};
    #pragma unroll
    for (int kk = 0; kk < 12; ++kk) {
      const int koff = kk * 32 + 8 * half;
      const v8bf lo = *(const v8bf*)(h_tile + l16 * HS + koff);
      const v8bf hi = *(const v8bf*)(h_tile + l16 * HS + koff + 16);
      const v16bf av = cat8(lo, hi);
      v16bf bv;
      #pragma unroll
      for (int j = 0; j < 8; ++j) {
        bv[j]     = (bf16)W1[(koff + j)      * GGDIM + col];
        bv[j + 8] = (bf16)W1[(koff + 16 + j) * GGDIM + col];
      }
      acc = __builtin_amdgcn_wmma_f32_16x16x32_bf16(false, av, false, bv, (short)0, acc, false, false);
    }
    const float b1c = b1[col];
    #pragma unroll
    for (int r = 0; r < 8; ++r) {
      const int row = half * 8 + r;
      float hv = acc[r] + b1c;
      hv = hv > 0.f ? hv : 0.01f * hv;
      h1_tile[row * H1S + col] = (bf16)hv;
    }
  }
  __syncthreads();

  // GEMM2: 16x256 @ 256x256.
  #pragma unroll
  for (int c = 0; c < 2; ++c) {
    const int col = (wave + c * 8) * 16 + l16;
    v8f acc = {0.f, 0.f, 0.f, 0.f, 0.f, 0.f, 0.f, 0.f};
    #pragma unroll
    for (int kk = 0; kk < 8; ++kk) {
      const int koff = kk * 32 + 8 * half;
      const v8bf lo = *(const v8bf*)(h1_tile + l16 * H1S + koff);
      const v8bf hi = *(const v8bf*)(h1_tile + l16 * H1S + koff + 16);
      const v16bf av = cat8(lo, hi);
      v16bf bv;
      #pragma unroll
      for (int j = 0; j < 8; ++j) {
        bv[j]     = (bf16)W2[(koff + j)      * GGDIM + col];
        bv[j + 8] = (bf16)W2[(koff + 16 + j) * GGDIM + col];
      }
      acc = __builtin_amdgcn_wmma_f32_16x16x32_bf16(false, av, false, bv, (short)0, acc, false, false);
    }
    const float b2c = b2[col];
    #pragma unroll
    for (int r = 0; r < 8; ++r) {
      const int row = half * 8 + r;
      const int gg  = g0 + row;
      if (gg < G) gfeat[(size_t)gg * GGDIM + col] = acc[r] + b2c;
    }
  }
}

// ---------------------------------------------------------------------------
extern "C" void kernel_launch(void* const* d_in, const int* in_sizes, int n_in,
                              void* d_out_v, int out_size, void* d_ws, size_t ws_size,
                              hipStream_t stream)
{
  (void)ws_size;
  const float* x[3] = {(const float*)d_in[0], (const float*)d_in[2], (const float*)d_in[4]};
  const int N = in_sizes[1];
  const int G = out_size / (3 * CDIM + GGDIM);

  const float *tok[3], *Wl[3], *bl[3], *Wr[3], *br[3], *att[3], *bo[3];
  const float *W1, *b1, *W2, *b2;

  if (n_in >= 31 && in_sizes[6] == CDIM * CDIM) {
    // jax tree_flatten (sorted keys): AGV, machine, mix, operation;
    // inner: Wl, Wr, att, bl, bo, br, tok ; mix: W1, W2, b1, b2
    const int base[3] = {24, 13, 6};   // operation, machine, AGV
    for (int t = 0; t < 3; ++t) {
      Wl[t]  = (const float*)d_in[base[t] + 0];
      Wr[t]  = (const float*)d_in[base[t] + 1];
      att[t] = (const float*)d_in[base[t] + 2];
      bl[t]  = (const float*)d_in[base[t] + 3];
      bo[t]  = (const float*)d_in[base[t] + 4];
      br[t]  = (const float*)d_in[base[t] + 5];
      tok[t] = (const float*)d_in[base[t] + 6];
    }
    W1 = (const float*)d_in[20]; W2 = (const float*)d_in[21];
    b1 = (const float*)d_in[22]; b2 = (const float*)d_in[23];
  } else {
    // insertion-order flatten: operation, machine, AGV {tok,Wl,bl,Wr,br,att,bo}, mix {W1,b1,W2,b2}
    const int base[3] = {6, 13, 20};
    for (int t = 0; t < 3; ++t) {
      tok[t] = (const float*)d_in[base[t] + 0];
      Wl[t]  = (const float*)d_in[base[t] + 1];
      bl[t]  = (const float*)d_in[base[t] + 2];
      Wr[t]  = (const float*)d_in[base[t] + 3];
      br[t]  = (const float*)d_in[base[t] + 4];
      att[t] = (const float*)d_in[base[t] + 5];
      bo[t]  = (const float*)d_in[base[t] + 6];
    }
    W1 = (const float*)d_in[27]; b1 = (const float*)d_in[28];
    W2 = (const float*)d_in[29]; b2 = (const float*)d_in[30];
  }

  float* out = (float*)d_out_v;
  float* gout[3] = {out, out + (size_t)G * CDIM, out + (size_t)2 * G * CDIM};
  float* gfeat   = out + (size_t)3 * G * CDIM;
  float* xd      = (float*)d_ws;   // 3*128 floats of scratch

  for (int t = 0; t < 3; ++t)
    xd_kernel<<<1, CDIM, 0, stream>>>(tok[t], Wr[t], br[t], xd + t * CDIM);
  for (int t = 0; t < 3; ++t)
    gat_kernel<<<G, 256, 0, stream>>>(x[t], Wl[t], bl[t], att[t], xd + t * CDIM, bo[t], gout[t], N, G);
  mix_kernel<<<(G + 15) / 16, 256, 0, stream>>>(gout[0], gout[1], gout[2],
                                                W1, b1, W2, b2, gfeat, G);
}
